// SoftKNN_9904194584582
// MI455X (gfx1250) — matrix operand order repaired and checked by reading, not compiled, not run on hardware
//
#include <hip/hip_runtime.h>
#include <stdint.h>

// ---------------------------------------------------------------------------
// Soft-KNN for MI455X (gfx1250, wave32, WMMA)
//   d2(q,n) = |x_q|^2 + |t_n|^2 - 2 x_q . t_n
//   GEMM via V_WMMA_F32_16X16X32_BF16 with split-bf16 error compensation:
//     x.t ~= xh.th + xh.tl + xl.th   (f32 accumulate)
//   Train features (bf16 hi+lo = 102 MB) are L2-resident (192 MB L2), so B
//   fragments are loaded straight from global in WMMA layout.
// ---------------------------------------------------------------------------

typedef __bf16 bf16_t;
typedef bf16_t bf16x8  __attribute__((ext_vector_type(8)));
typedef bf16_t bf16x16 __attribute__((ext_vector_type(16)));
typedef float  f32x8   __attribute__((ext_vector_type(8)));

#define DIMS   512
#define NTRAIN 50000
#define NQ     2048
#define NCLASS 100
#define KNN    16
#define NPART  4
#define NPER   (NTRAIN / NPART)      // 12500
#define QB     64                    // queries per block (4 row tiles)
#define NT     256                   // candidates per chunk (16 col tiles)
#define NCHUNK ((NPER + NT - 1) / NT)
#define LDA    520                   // A row stride in LDS (bf16 elems, +8 pad)
#define LDD    260                   // dist row stride in LDS (f32 elems, +4 pad)

__device__ __forceinline__ unsigned short f32_to_bf16_rne(float f) {
  unsigned u = __float_as_uint(f);
  u += 0x7FFFu + ((u >> 16) & 1u);
  return (unsigned short)(u >> 16);
}
__device__ __forceinline__ int imin(int a, int b) { return a < b ? a : b; }

// ---------------------------------------------------------------------------
// Kernel 1: split every row (train + query) into bf16 hi/lo and compute norms.
// One 128-thread block per row; rows [0,50000) = train, [50000,52048) = query.
// ---------------------------------------------------------------------------
__global__ __launch_bounds__(128)
void prep_split(const float* __restrict__ x, const float* __restrict__ tf,
                unsigned short* __restrict__ thi, unsigned short* __restrict__ tlo,
                unsigned short* __restrict__ xhi, unsigned short* __restrict__ xlo,
                float* __restrict__ ynorm, float* __restrict__ xnorm) {
  const int row = blockIdx.x;
  const int t   = threadIdx.x;
  const float* src; unsigned short* dhi; unsigned short* dlo; float* nrm;
  if (row < NTRAIN) {
    src = tf + (size_t)row * DIMS;
    dhi = thi + (size_t)row * DIMS; dlo = tlo + (size_t)row * DIMS;
    nrm = ynorm + row;
  } else {
    const int r2 = row - NTRAIN;
    src = x + (size_t)r2 * DIMS;
    dhi = xhi + (size_t)r2 * DIMS; dlo = xlo + (size_t)r2 * DIMS;
    nrm = xnorm + r2;
  }
  float ss = 0.f;
  for (int i = t; i < DIMS; i += 128) {
    const float f = src[i];
    const unsigned short h = f32_to_bf16_rne(f);
    const float hf = __uint_as_float((unsigned)h << 16);
    const unsigned short l = f32_to_bf16_rne(f - hf);
    dhi[i] = h; dlo[i] = l;
    ss += f * f;
  }
#pragma unroll
  for (int m = 16; m >= 1; m >>= 1) ss += __shfl_xor(ss, m, 32);
  __shared__ float red[4];
  if ((t & 31) == 0) red[t >> 5] = ss;
  __syncthreads();
  if (t == 0) *nrm = red[0] + red[1] + red[2] + red[3];
}

// ---------------------------------------------------------------------------
// Kernel 2: fused GEMM (WMMA bf16x3) + streaming top-16 per query, per
// N-partition. Grid = (NQ/QB, NPART), block = 512 threads = 16 waves.
//   wave w: rowTile = w&3 (16 queries), colGrp = w>>2 (64 candidates)
// ---------------------------------------------------------------------------
__global__ __launch_bounds__(512, 1)
void knn_main(const unsigned short* __restrict__ xhi, const unsigned short* __restrict__ xlo,
              const unsigned short* __restrict__ thi, const unsigned short* __restrict__ tlo,
              const float* __restrict__ xnorm, const float* __restrict__ ynorm,
              float* __restrict__ pd2, int* __restrict__ pidx) {
  extern __shared__ char smem[];
  unsigned short* sAhi = (unsigned short*)smem;                       // 66,560 B
  unsigned short* sAlo = (unsigned short*)(smem + QB * LDA * 2);      // 66,560 B
  float* sDot = (float*)(smem + 2 * QB * LDA * 2);                    // 66,560 B
  float* sXn  = (float*)(smem + 2 * QB * LDA * 2 + QB * LDD * 4);     //    256 B

  const int tid  = threadIdx.x;
  const int lane = tid & 31;
  const int wv   = tid >> 5;
  const int q0   = blockIdx.x * QB;
  const int n0   = blockIdx.y * NPER;
  const int nEnd = n0 + NPER;

  // ---- stage query tile (bf16 hi/lo) + norms into LDS ----
  for (int c = tid; c < QB * (DIMS / 8); c += 512) {
    const int row = c >> 6, k8 = c & 63;
    *(bf16x8*)(sAhi + row * LDA + k8 * 8) =
        *(const bf16x8*)(xhi + (size_t)(q0 + row) * DIMS + k8 * 8);
    *(bf16x8*)(sAlo + row * LDA + k8 * 8) =
        *(const bf16x8*)(xlo + (size_t)(q0 + row) * DIMS + k8 * 8);
  }
  if (tid < QB) sXn[tid] = xnorm[q0 + tid];
  __syncthreads();

  const int rowTile = wv & 3;
  const int colGrp  = wv >> 2;
  const int lcol    = lane & 15;  // WMMA fragment column / A row
  const int lhalf   = lane >> 4;  // WMMA fragment K-half

  // top-k scan assignment: 8 lanes cooperate on each query row
  const int srow = (wv << 2) + (lane >> 3);  // 0..63 (query-local row)
  const int sub  = lane & 7;                 // 0..7  (sub-lane within row)
  float ld2[KNN]; int lidx[KNN];
#pragma unroll
  for (int j = 0; j < KNN; ++j) { ld2[j] = __builtin_inff(); lidx[j] = 0; }

  const int arow = rowTile * 16 + lcol;

  for (int ch = 0; ch < NCHUNK; ++ch) {
    const int cb = n0 + ch * NT;

    f32x8 acc[4];
#pragma unroll
    for (int ct = 0; ct < 4; ++ct) acc[ct] = (f32x8)0.f;

    int candc[4]; size_t boff[4];
#pragma unroll
    for (int ct = 0; ct < 4; ++ct) {
      const int cand = cb + colGrp * 64 + ct * 16 + lcol;
      candc[ct] = cand;
      boff[ct]  = (size_t)imin(cand, nEnd - 1) * DIMS + lhalf * 16;
    }

    // ---- K loop: 16 steps of K=32, 3 compensated WMMAs per tile ----
#pragma unroll 4
    for (int ks = 0; ks < 16; ++ks) {
      const int k0 = ks * 32;
      union { bf16x16 v; bf16x8 h[2]; } ah, al;
      ah.h[0] = *(const bf16x8*)(sAhi + arow * LDA + k0 + lhalf * 16);
      ah.h[1] = *(const bf16x8*)(sAhi + arow * LDA + k0 + lhalf * 16 + 8);
      al.h[0] = *(const bf16x8*)(sAlo + arow * LDA + k0 + lhalf * 16);
      al.h[1] = *(const bf16x8*)(sAlo + arow * LDA + k0 + lhalf * 16 + 8);
#pragma unroll
      for (int ct = 0; ct < 4; ++ct) {
        const bf16x16 bh = *(const bf16x16*)(thi + boff[ct] + k0);
        const bf16x16 bl = *(const bf16x16*)(tlo + boff[ct] + k0);
        acc[ct] = __builtin_amdgcn_wmma_f32_16x16x32_bf16(
            false, ah.v, false, bh, (short)0, acc[ct], false, false);
        acc[ct] = __builtin_amdgcn_wmma_f32_16x16x32_bf16(
            false, ah.v, false, bl, (short)0, acc[ct], false, false);
        acc[ct] = __builtin_amdgcn_wmma_f32_16x16x32_bf16(
            false, al.v, false, bh, (short)0, acc[ct], false, false);
      }
    }

    // ---- epilogue: d2 = |x|^2 + |t|^2 - 2*dot, into LDS dist tile ----
#pragma unroll
    for (int ct = 0; ct < 4; ++ct) {
      const int cand   = candc[ct];
      const bool valid = cand < nEnd;
      const float yn   = ynorm[valid ? cand : nEnd - 1];
      const int gcol   = colGrp * 64 + ct * 16 + lcol;
#pragma unroll
      for (int v = 0; v < 8; ++v) {
        const int grow = rowTile * 16 + (lhalf << 3) + v;  // C layout: M = v + 8*half
        const float d2 = fmaxf(sXn[grow] + yn - 2.0f * acc[ct][v], 0.0f);
        sDot[grow * LDD + gcol] = valid ? d2 : __builtin_inff();
      }
      // hint next chunk's candidate row toward L0/L2
      __builtin_prefetch(thi + (size_t)imin(cand + NT, nEnd - 1) * DIMS + lhalf * 16, 0, 1);
    }
    __syncthreads();

    // ---- streaming top-16 update (threshold-gated sorted insert) ----
#pragma unroll 8
    for (int j = 0; j < NT / 8; ++j) {
      const int col  = sub + (j << 3);
      const float d2 = sDot[srow * LDD + col];
      if (d2 < ld2[KNN - 1]) {
        ld2[KNN - 1] = d2; lidx[KNN - 1] = cb + col;
#pragma unroll
        for (int t = KNN - 1; t > 0; --t) {
          if (ld2[t] < ld2[t - 1]) {
            const float fd = ld2[t]; ld2[t] = ld2[t - 1]; ld2[t - 1] = fd;
            const int   ti = lidx[t]; lidx[t] = lidx[t - 1]; lidx[t - 1] = ti;
          }
        }
      }
    }
    __syncthreads();
  }

  // ---- merge 8 sorted lane-lists -> exact top-16 per row (reuse A LDS) ----
  float* md2 = (float*)smem;                              // 64*128 f32 = 32 KB
  int*   mix = (int*)(smem + QB * 128 * sizeof(float));   // 64*128 i32 = 32 KB
#pragma unroll
  for (int j = 0; j < KNN; ++j) {
    md2[srow * 128 + sub * KNN + j] = ld2[j];
    mix[srow * 128 + sub * KNN + j] = lidx[j];
  }
  __syncthreads();

  int p = 0;
  for (int r = 0; r < KNN; ++r) {
    const float hv = (p < KNN) ? md2[srow * 128 + sub * KNN + p] : __builtin_inff();
    unsigned long long key = ((unsigned long long)__float_as_uint(hv) << 32) |
                             (unsigned long long)((sub << 5) | p);
#pragma unroll
    for (int m = 1; m < 8; m <<= 1) {
      const unsigned long long o = __shfl_xor(key, m, 8);
      key = (o < key) ? o : key;
    }
    const int wsub = (int)((key >> 5) & 7u);
    const int wp   = (int)(key & 31u);
    if (sub == wsub) ++p;
    if (sub == 0) {
      const size_t o = ((size_t)blockIdx.y * NQ + (q0 + srow)) * KNN + r;
      pd2[o]  = __uint_as_float((unsigned)(key >> 32));
      pidx[o] = mix[srow * 128 + wsub * KNN + wp];
    }
  }
}

// ---------------------------------------------------------------------------
// Kernel 3: merge 4 partition lists per query, softmax over 16 neighbor
// distances, scatter into 100 class buckets (LDS ds_add_f32), write row.
// Block = 128 threads = 4 waves; one query per wave.
// ---------------------------------------------------------------------------
__global__ __launch_bounds__(128, 1)
void knn_finalize(const float* __restrict__ pd2, const int* __restrict__ pidx,
                  const int* __restrict__ labels, float* __restrict__ out) {
  const int tid  = threadIdx.x;
  const int lane = tid & 31;
  const int wv   = tid >> 5;
  const int q    = blockIdx.x * 4 + wv;

  const int e0 = lane, e1 = lane + 32;  // 64 candidate entries, 2 per lane
  const size_t o0 = ((size_t)(e0 >> 4) * NQ + q) * KNN + (e0 & 15);
  const size_t o1 = ((size_t)(e1 >> 4) * NQ + q) * KNN + (e1 & 15);
  float d0 = pd2[o0]; int i0 = pidx[o0];
  float d1 = pd2[o1]; int i1 = pidx[o1];
  bool c0 = false, c1 = false;
  float selD = 0.f; int selI = 0;

  for (int r = 0; r < KNN; ++r) {
    const unsigned long long k0 = c0 ? ~0ull
        : (((unsigned long long)__float_as_uint(d0) << 32) |
           ((unsigned long long)e0 << 24) | (unsigned)i0);
    const unsigned long long k1 = c1 ? ~0ull
        : (((unsigned long long)__float_as_uint(d1) << 32) |
           ((unsigned long long)e1 << 24) | (unsigned)i1);
    unsigned long long key = (k0 < k1) ? k0 : k1;
#pragma unroll
    for (int m = 1; m < 32; m <<= 1) {
      const unsigned long long o = __shfl_xor(key, m, 32);
      key = (o < key) ? o : key;
    }
    const int owner = (int)((key >> 24) & 0xFFu);
    if (owner == e0) c0 = true;
    if (owner == e1) c1 = true;
    if (lane == r) { selD = __uint_as_float((unsigned)(key >> 32));
                     selI = (int)(key & 0xFFFFFFu); }
  }

  // softmax over -dist (TEMPERATURE = 1); lane 0 holds min distance
  const float dist = sqrtf(fmaxf(selD, 0.f));
  const float dmin = __shfl(dist, 0, 32);
  float w = (lane < KNN) ? __expf(dmin - dist) : 0.f;
  float s = w;
#pragma unroll
  for (int m = 1; m < 32; m <<= 1) s += __shfl_xor(s, m, 32);
  w /= s;

  __shared__ float buck[4][112];
  for (int c = lane; c < 112; c += 32) buck[wv][c] = 0.f;
  __syncthreads();
  if (lane < KNN) atomicAdd(&buck[wv][labels[selI]], w);
  __syncthreads();
  for (int c = lane; c < NCLASS; c += 32) out[(size_t)q * NCLASS + c] = buck[wv][c];
}

// ---------------------------------------------------------------------------
extern "C" void kernel_launch(void* const* d_in, const int* in_sizes, int n_in,
                              void* d_out, int out_size, void* d_ws, size_t ws_size,
                              hipStream_t stream) {
  (void)in_sizes; (void)n_in; (void)out_size; (void)ws_size;
  const float* x      = (const float*)d_in[0];
  const float* tf     = (const float*)d_in[1];
  const int*   labels = (const int*)d_in[2];

  char* ws = (char*)d_ws;
  size_t off = 0;
  auto take = [&](size_t bytes) -> void* {
    void* p = ws + off;
    off = (off + bytes + 255) & ~(size_t)255;
    return p;
  };
  unsigned short* t_hi = (unsigned short*)take((size_t)NTRAIN * DIMS * 2);  // 51.2 MB
  unsigned short* t_lo = (unsigned short*)take((size_t)NTRAIN * DIMS * 2);  // 51.2 MB
  unsigned short* x_hi = (unsigned short*)take((size_t)NQ * DIMS * 2);      //  2.1 MB
  unsigned short* x_lo = (unsigned short*)take((size_t)NQ * DIMS * 2);      //  2.1 MB
  float* y_norm = (float*)take((size_t)NTRAIN * 4);
  float* x_norm = (float*)take((size_t)NQ * 4);
  float* p_d2   = (float*)take((size_t)NPART * NQ * KNN * 4);
  int*   p_idx  = (int*)take((size_t)NPART * NQ * KNN * 4);

  prep_split<<<NTRAIN + NQ, 128, 0, stream>>>(x, tf, t_hi, t_lo, x_hi, x_lo,
                                              y_norm, x_norm);

  const size_t lds = (size_t)(2 * QB * LDA * 2 + QB * LDD * 4 + QB * 4);  // 199,936 B
  knn_main<<<dim3(NQ / QB, NPART), 512, lds, stream>>>(
      x_hi, x_lo, t_hi, t_lo, x_norm, y_norm, p_d2, p_idx);

  knn_finalize<<<NQ / 4, 128, 0, stream>>>(p_d2, p_idx, labels, (float*)d_out);
}
